// EpGRUCell_35064113004713
// MI455X (gfx1250) — compile-verified, matrix operand order checked by reading, not run on hardware
//
#include <hip/hip_runtime.h>
#include <hip/hip_bf16.h>
#include <math.h>

// ---------------- problem constants (from reference) ----------------
#define T_STEPS 512
#define BATCH   64
#define DX      1024
#define DH      1024
#define TB      (T_STEPS * BATCH)      // 32768 rows in big GEMM
#define NCAT    (4 * DH)               // 4096 = 3*DH gates + DH candidate

// ---------------- types for CDNA5 WMMA ----------------
typedef __attribute__((ext_vector_type(16))) __bf16       v16bf;
typedef __attribute__((ext_vector_type(8)))  float        v8f;
typedef __attribute__((ext_vector_type(4)))  unsigned int u32x4;

__device__ __forceinline__ unsigned short f2bf(float f) {
  unsigned u = __float_as_uint(f);
  u += 0x7FFFu + ((u >> 16) & 1u);     // round-to-nearest-even
  return (unsigned short)(u >> 16);
}
__device__ __forceinline__ float bf2f(unsigned short s) {
  return __uint_as_float(((unsigned)s) << 16);
}

// Load a 16x32 bf16 K-major fragment per the CDNA5 ISA layout (CDNA5 ISA):
// lanes 0-15: rows r0..r0+15, K = k0+0..7 (VGPR0-3) and k0+16..23 (VGPR4-7)
// lanes 16-31: same rows,     K = k0+8..15            and k0+24..31
// Works for A (M x K row-major) and for B given as W = B^T (N x K row-major).
union FragU { u32x4 q[2]; v16bf v; };
__device__ __forceinline__ v16bf load_kmajor_frag(const unsigned short* __restrict__ base,
                                                  int ld, int r0, int k0) {
  int lane = threadIdx.x & 31;
  int r    = lane & 15;
  int h    = lane >> 4;                // 0 or 1 -> K sub-chunk select
  const unsigned short* p = base + (size_t)(r0 + r) * ld + (k0 + h * 8);
  FragU f;
  f.q[0] = *(const u32x4*)(p);         // K = k0+h*8   .. +7   (16B)
  f.q[1] = *(const u32x4*)(p + 16);    // K = k0+16+h*8 .. +7  (16B)
  return f.v;
}

#define WMMA_BF16(A, Bm, C) \
  __builtin_amdgcn_wmma_f32_16x16x32_bf16(false, (A), false, (Bm), (short)0, (C), false, false)

// ---------------- elementwise helpers ----------------
__global__ __launch_bounds__(256) void k_f32_to_bf16(const float* __restrict__ src,
                                                     unsigned short* __restrict__ dst,
                                                     size_t n) {
  size_t i = (size_t)blockIdx.x * 256 + threadIdx.x;
  if (i < n) dst[i] = f2bf(src[i]);
}

__global__ __launch_bounds__(256) void k_copy_f32(const float* __restrict__ src,
                                                  float* __restrict__ dst, int n) {
  int i = blockIdx.x * 256 + threadIdx.x;
  if (i < n) dst[i] = src[i];
}

// ---------------- big fused input projection ----------------
// C[32768,4096](bf16) = A[32768,1024](bf16) x Wcat[4096,1024]^T + [b_ih | b_xh]
// One wave computes a 32x64 strip: 2 M-tiles x 4 N-tiles = 8 WMMAs per k-step
// from 2 A-fragments + 4 B-fragments. Fragment loads are double-buffered so
// next-k loads are in flight while current-k WMMAs execute.
__global__ __launch_bounds__(256) void gemm_xproj(
    const unsigned short* __restrict__ A,     // xs  bf16 [TB, DX]
    const unsigned short* __restrict__ Bw,    // Wcat bf16 [NCAT, DX]
    const float* __restrict__ b_ih,           // [3*DH]
    const float* __restrict__ b_xh,           // [DH]
    unsigned short* __restrict__ C)           // xgh bf16 [TB, NCAT]
{
  const int NG = NCAT / 64;                   // 64 n-groups
  int waveId = blockIdx.x * 8 + (threadIdx.x >> 5);
  int m0 = (waveId / NG) * 32;
  int n0 = (waveId % NG) * 64;

  v8f acc[2][4] = {};

  // prologue: fragments for k=0
  v16bf a0 = load_kmajor_frag(A, DX, m0,      0);
  v16bf a1 = load_kmajor_frag(A, DX, m0 + 16, 0);
  v16bf b0 = load_kmajor_frag(Bw, DX, n0,      0);
  v16bf b1 = load_kmajor_frag(Bw, DX, n0 + 16, 0);
  v16bf b2 = load_kmajor_frag(Bw, DX, n0 + 32, 0);
  v16bf b3 = load_kmajor_frag(Bw, DX, n0 + 48, 0);

  for (int k0 = 0; k0 < DX; k0 += 32) {
    int kn = (k0 + 32 < DX) ? k0 + 32 : k0;   // last iter: reload (dead) same k
    // stream the A panel ahead through GL2 (global_prefetch_b8)
    __builtin_prefetch(A + (size_t)m0 * DX + kn + 32, 0, 1);
    // issue next-iteration loads BEFORE consuming current fragments
    v16bf na0 = load_kmajor_frag(A, DX, m0,      kn);
    v16bf na1 = load_kmajor_frag(A, DX, m0 + 16, kn);
    v16bf nb0 = load_kmajor_frag(Bw, DX, n0,      kn);
    v16bf nb1 = load_kmajor_frag(Bw, DX, n0 + 16, kn);
    v16bf nb2 = load_kmajor_frag(Bw, DX, n0 + 32, kn);
    v16bf nb3 = load_kmajor_frag(Bw, DX, n0 + 48, kn);

    acc[0][0] = WMMA_BF16(a0, b0, acc[0][0]);
    acc[0][1] = WMMA_BF16(a0, b1, acc[0][1]);
    acc[0][2] = WMMA_BF16(a0, b2, acc[0][2]);
    acc[0][3] = WMMA_BF16(a0, b3, acc[0][3]);
    acc[1][0] = WMMA_BF16(a1, b0, acc[1][0]);
    acc[1][1] = WMMA_BF16(a1, b1, acc[1][1]);
    acc[1][2] = WMMA_BF16(a1, b2, acc[1][2]);
    acc[1][3] = WMMA_BF16(a1, b3, acc[1][3]);

    a0 = na0; a1 = na1;
    b0 = nb0; b1 = nb1; b2 = nb2; b3 = nb3;
  }

  int lane = threadIdx.x & 31;
  #pragma unroll
  for (int mi = 0; mi < 2; ++mi) {
    int rbase = m0 + mi * 16 + ((lane >> 4) << 3);  // C layout: VGPR j -> row rbase+j
    #pragma unroll
    for (int i = 0; i < 4; ++i) {
      int col    = n0 + i * 16 + (lane & 15);
      float bias = (col < 3 * DH) ? b_ih[col] : b_xh[col - 3 * DH];
      #pragma unroll
      for (int j = 0; j < 8; ++j)
        C[(size_t)(rbase + j) * NCAT + col] = f2bf(acc[mi][i][j] + bias);
    }
  }
}

// ---------------- scan step 1: gates ----------------
// hg = h_bf x W_hh^T ; g = sigmoid(xg_t + hg + b_hh)
// n0 <  DH : r gate -> rh_bf = bf16(r * h_prev)
// n0 < 2DH : z gate -> zbuf ; else e gate -> ebuf
// 1 wave per 16x16 tile: maximum parallelism in the latency-bound scan.
__global__ __launch_bounds__(256) void gru_gates(
    const unsigned short* __restrict__ Hbf,   // [BATCH, DH] bf16
    const unsigned short* __restrict__ Whh,   // [3*DH, DH] bf16
    const float* __restrict__ b_hh,           // [3*DH]
    const unsigned short* __restrict__ xg_t,  // xgh + t*BATCH*NCAT, [BATCH, NCAT]
    const float* __restrict__ h_prev,         // [BATCH, DH] f32
    unsigned short* __restrict__ rh_bf,       // [BATCH, DH] bf16
    float* __restrict__ zbuf,                 // [BATCH, DH]
    float* __restrict__ ebuf)                 // [BATCH, DH]
{
  const int NT = (3 * DH) / 16;               // 192 n-tiles
  int waveId = blockIdx.x * 8 + (threadIdx.x >> 5);   // 0..767
  int m0 = (waveId / NT) * 16;
  int n0 = (waveId % NT) * 16;

  v8f acc = {};
  v16bf a = load_kmajor_frag(Hbf, DH, m0, 0);
  v16bf b = load_kmajor_frag(Whh, DH, n0, 0);
  for (int k0 = 0; k0 < DH; k0 += 32) {
    int kn = (k0 + 32 < DH) ? k0 + 32 : k0;
    v16bf an = load_kmajor_frag(Hbf, DH, m0, kn);
    v16bf bn = load_kmajor_frag(Whh, DH, n0, kn);
    acc = WMMA_BF16(a, b, acc);
    a = an; b = bn;
  }

  int lane  = threadIdx.x & 31;
  int col   = n0 + (lane & 15);
  int rbase = m0 + ((lane >> 4) << 3);
  float bias = b_hh[col];
  #pragma unroll
  for (int j = 0; j < 8; ++j) {
    int row = rbase + j;
    float x = acc[j] + bias + bf2f(xg_t[(size_t)row * NCAT + col]);
    float g = 1.0f / (1.0f + __expf(-x));
    if (n0 < DH) {
      size_t idx = (size_t)row * DH + col;
      rh_bf[idx] = f2bf(g * h_prev[idx]);
    } else if (n0 < 2 * DH) {
      zbuf[(size_t)row * DH + (col - DH)] = g;
    } else {
      ebuf[(size_t)row * DH + (col - 2 * DH)] = g;
    }
  }
}

// ---------------- scan step 2: candidate + state update ----------------
// hh = tanh((r*h) x W_r^T + b_r + xh_t) ; h = z*h + (1-z)*hh + e*tanh(ms_t)
__global__ __launch_bounds__(256) void gru_update(
    const unsigned short* __restrict__ RHbf,  // [BATCH, DH] bf16
    const unsigned short* __restrict__ Wr,    // [DH, DH] bf16
    const float* __restrict__ b_r,            // [DH]
    const unsigned short* __restrict__ xgh_t, // [BATCH, NCAT]; xh at cols 3*DH..
    const float* __restrict__ zbuf,
    const float* __restrict__ ebuf,
    const float* __restrict__ h_prev,         // [BATCH, DH] f32
    const float* __restrict__ ms_t,           // [BATCH, DH] f32
    float* __restrict__ h_out,                // d_out + t*BATCH*DH
    unsigned short* __restrict__ Hbf)         // next-step bf16 h
{
  const int NT = DH / 16;                     // 64 n-tiles
  int waveId = blockIdx.x * 8 + (threadIdx.x >> 5);   // 0..255
  int m0 = (waveId / NT) * 16;
  int n0 = (waveId % NT) * 16;

  v8f acc = {};
  v16bf a = load_kmajor_frag(RHbf, DH, m0, 0);
  v16bf b = load_kmajor_frag(Wr,   DH, n0, 0);
  for (int k0 = 0; k0 < DH; k0 += 32) {
    int kn = (k0 + 32 < DH) ? k0 + 32 : k0;
    v16bf an = load_kmajor_frag(RHbf, DH, m0, kn);
    v16bf bn = load_kmajor_frag(Wr,   DH, n0, kn);
    acc = WMMA_BF16(a, b, acc);
    a = an; b = bn;
  }

  int lane  = threadIdx.x & 31;
  int col   = n0 + (lane & 15);
  int rbase = m0 + ((lane >> 4) << 3);
  float bias = b_r[col];
  #pragma unroll
  for (int j = 0; j < 8; ++j) {
    int row = rbase + j;
    size_t idx = (size_t)row * DH + col;
    float pre = acc[j] + bias + bf2f(xgh_t[(size_t)row * NCAT + 3 * DH + col]);
    float hh  = tanhf(pre);
    float z   = zbuf[idx];
    float e   = ebuf[idx];
    float hp  = h_prev[idx];
    float tm  = tanhf(ms_t[idx]);
    float ht  = z * hp + (1.0f - z) * hh + e * tm;
    h_out[idx] = ht;
    Hbf[idx]   = f2bf(ht);
  }
}

// ---------------- host-side orchestration ----------------
extern "C" void kernel_launch(void* const* d_in, const int* in_sizes, int n_in,
                              void* d_out, int out_size, void* d_ws, size_t ws_size,
                              hipStream_t stream) {
  const float* xs   = (const float*)d_in[0];
  const float* ms   = (const float*)d_in[1];
  const float* h0   = (const float*)d_in[2];
  const float* W_ih = (const float*)d_in[3];
  const float* b_ih = (const float*)d_in[4];
  const float* W_hh = (const float*)d_in[5];
  const float* b_hh = (const float*)d_in[6];
  const float* W_xh = (const float*)d_in[7];
  const float* b_xh = (const float*)d_in[8];
  const float* W_r  = (const float*)d_in[9];
  const float* b_r  = (const float*)d_in[10];
  float* out = (float*)d_out;                 // [T,B,DH] outs ++ [B,DH] h_last

  // workspace carve-out (256B aligned slices)
  char* ws = (char*)d_ws;
  size_t off = 0;
  auto carve = [&](size_t bytes) -> void* {
    void* p = ws + off;
    off = (off + bytes + 255) & ~(size_t)255;
    return p;
  };
  unsigned short* xs_bf   = (unsigned short*)carve((size_t)TB * DX * 2);      // 64 MiB
  unsigned short* Wcat_bf = (unsigned short*)carve((size_t)NCAT * DX * 2);    //  8 MiB
  unsigned short* Whh_bf  = (unsigned short*)carve((size_t)3 * DH * DH * 2);  //  6 MiB
  unsigned short* Wr_bf   = (unsigned short*)carve((size_t)DH * DH * 2);      //  2 MiB
  unsigned short* xgh     = (unsigned short*)carve((size_t)TB * NCAT * 2);    // 256 MiB
  unsigned short* rh_bf   = (unsigned short*)carve((size_t)BATCH * DH * 2);
  unsigned short* h_bf    = (unsigned short*)carve((size_t)BATCH * DH * 2);
  float*          zbuf    = (float*)carve((size_t)BATCH * DH * 4);
  float*          ebuf    = (float*)carve((size_t)BATCH * DH * 4);
  (void)ws_size; (void)in_sizes; (void)n_in; (void)out_size;

  auto cdiv = [](size_t a, size_t b) { return (unsigned)((a + b - 1) / b); };

  // 1) pack everything to bf16 (weights already N x K = K-major for WMMA-B)
  k_f32_to_bf16<<<cdiv((size_t)TB * DX, 256), 256, 0, stream>>>(xs, xs_bf, (size_t)TB * DX);
  k_f32_to_bf16<<<cdiv((size_t)3 * DH * DX, 256), 256, 0, stream>>>(W_ih, Wcat_bf, (size_t)3 * DH * DX);
  k_f32_to_bf16<<<cdiv((size_t)DH * DX, 256), 256, 0, stream>>>(W_xh, Wcat_bf + (size_t)3 * DH * DX, (size_t)DH * DX);
  k_f32_to_bf16<<<cdiv((size_t)3 * DH * DH, 256), 256, 0, stream>>>(W_hh, Whh_bf, (size_t)3 * DH * DH);
  k_f32_to_bf16<<<cdiv((size_t)DH * DH, 256), 256, 0, stream>>>(W_r, Wr_bf, (size_t)DH * DH);
  k_f32_to_bf16<<<cdiv((size_t)BATCH * DH, 256), 256, 0, stream>>>(h0, h_bf, (size_t)BATCH * DH);

  // 2) fused hoisted projections: xgh = xs x [W_ih;W_xh]^T + bias   (bf16 out)
  {
    unsigned waves = (TB / 32) * (NCAT / 64);           // 65536 waves, 32x64/wave
    gemm_xproj<<<waves / 8, 256, 0, stream>>>(xs_bf, Wcat_bf, b_ih, b_xh, xgh);
  }

  // 3) sequential scan: 2 fused WMMA kernels per step
  for (int t = 0; t < T_STEPS; ++t) {
    const float* h_prev = (t == 0) ? h0 : (out + (size_t)(t - 1) * BATCH * DH);
    const unsigned short* xgh_t = xgh + (size_t)t * BATCH * NCAT;
    const float* ms_t = ms + (size_t)t * BATCH * DH;
    float* h_out = out + (size_t)t * BATCH * DH;

    gru_gates<<<((BATCH / 16) * (3 * DH / 16)) / 8, 256, 0, stream>>>(
        h_bf, Whh_bf, b_hh, xgh_t, h_prev, rh_bf, zbuf, ebuf);
    gru_update<<<((BATCH / 16) * (DH / 16)) / 8, 256, 0, stream>>>(
        rh_bf, Wr_bf, b_r, xgh_t, zbuf, ebuf, h_prev, ms_t, h_out, h_bf);
  }

  // 4) h_last = outs[T-1]
  k_copy_f32<<<(BATCH * DH) / 256, 256, 0, stream>>>(
      out + (size_t)(T_STEPS - 1) * BATCH * DH, out + (size_t)T_STEPS * BATCH * DH,
      BATCH * DH);
}
